// Transformer_24395414241461
// MI455X (gfx1250) — matrix operand order repaired and checked by reading, not compiled, not run on hardware
//
#include <hip/hip_runtime.h>
#include <math.h>

// ---------------- problem constants ----------------
#define S_   4096
#define D_   512
#define H_   8
#define DK_  64
#define DFF_ 2048
#define EPS_ 1e-5f

// ---------------- vector types for WMMA ----------------
typedef __attribute__((ext_vector_type(8)))  __bf16 v8bf;
typedef __attribute__((ext_vector_type(16))) __bf16 v16bf;
typedef __attribute__((ext_vector_type(8)))  float  v8f;

static __device__ inline v8bf ld8(const __bf16* p) { return *(const v8bf*)p; }
static __device__ inline v16bf cat16(v8bf a, v8bf b) {
  return __builtin_shufflevector(a, b, 0,1,2,3,4,5,6,7,8,9,10,11,12,13,14,15);
}
static __device__ inline v8f wmma_bf16(v16bf a, v16bf b, v8f c) {
  // D = A(16x32 bf16) * B(32x16 bf16) + C(16x16 f32)
  return __builtin_amdgcn_wmma_f32_16x16x32_bf16(false, a, false, b, (short)0, c, false, false);
}

// ============================================================
// 1) x = E + positional encoding; emit f32 and bf16 copies
// ============================================================
__global__ void posenc_kernel(const float* __restrict__ E,
                              float* __restrict__ x, __bf16* __restrict__ xb) {
  int idx = blockIdx.x * blockDim.x + threadIdx.x;   // over S_*D_
  int i = idx & (D_ - 1);
  int s = idx >> 9;                                  // D_ == 512
  float ex  = (float)(i & ~1) / (float)D_;
  float ang = (float)s * powf(10000.0f, -ex);
  float pe  = (i & 1) ? cosf(ang) : sinf(ang);
  float v = E[idx] + pe;
  x[idx]  = v;
  xb[idx] = (__bf16)v;
}

// ============================================================
// 2) transpose + convert weights: in f32 (R x C) -> out bf16 (C x R)
//    batched over blockIdx.z
// ============================================================
__global__ void transconv_kernel(const float* __restrict__ in, __bf16* __restrict__ out,
                                 int R, int C) {
  size_t base = (size_t)blockIdx.z * (size_t)R * (size_t)C;
  int idx = blockIdx.x * blockDim.x + threadIdx.x;
  if (idx >= R * C) return;
  int r = idx / C, c = idx - r * C;
  out[base + (size_t)c * R + r] = (__bf16)in[base + idx];
}

// ============================================================
// 3) generic bf16 WMMA GEMM:  C(MxN) = A(MxK) * Bt(NxK)^T
//    8 waves/block; each wave computes a 32x32 tile (2x2 WMMA tiles).
//    Epilogue: optional bias[col], residual[row,col], relu,
//              f32 out and/or bf16 out (optionally transposed: Cb[col*M+row]).
// ============================================================
__global__ void gemm_bf16_wmma(const __bf16* __restrict__ A, const __bf16* __restrict__ Bt,
                               int M, int N, int K,
                               float* __restrict__ Cf, __bf16* __restrict__ Cb, int cbT,
                               const float* __restrict__ bias,
                               const float* __restrict__ residual, int relu) {
  int lane = threadIdx.x & 31;
  int wave = threadIdx.x >> 5;
  int mBase = (blockIdx.y * 8 + wave) * 32;
  int nBase = blockIdx.x * 32;
  if (mBase >= M) return;
  int half = lane >> 4;      // 0: lanes 0-15, 1: lanes 16-31
  int l16  = lane & 15;
  int aOff = half ? 8 : 0;   // A fragment: K chunks {aOff..aOff+7, aOff+16..aOff+23}
  int bOff = half ? 8 : 0;   // B fragment: K chunks {bOff..bOff+7, bOff+16..bOff+23}

  const __bf16* aRow0 = A + (size_t)(mBase + l16) * K;
  const __bf16* aRow1 = aRow0 + (size_t)16 * K;
  const __bf16* bRow0 = Bt + (size_t)(nBase + l16) * K;
  const __bf16* bRow1 = bRow0 + (size_t)16 * K;

  v8f acc[2][2] = {};
  for (int k0 = 0; k0 < K; k0 += 32) {
    v16bf a0 = cat16(ld8(aRow0 + k0 + aOff), ld8(aRow0 + k0 + aOff + 16));
    v16bf a1 = cat16(ld8(aRow1 + k0 + aOff), ld8(aRow1 + k0 + aOff + 16));
    v16bf b0 = cat16(ld8(bRow0 + k0 + bOff), ld8(bRow0 + k0 + bOff + 16));
    v16bf b1 = cat16(ld8(bRow1 + k0 + bOff), ld8(bRow1 + k0 + bOff + 16));
    acc[0][0] = wmma_bf16(a0, b0, acc[0][0]);
    acc[0][1] = wmma_bf16(a0, b1, acc[0][1]);
    acc[1][0] = wmma_bf16(a1, b0, acc[1][0]);
    acc[1][1] = wmma_bf16(a1, b1, acc[1][1]);
  }

#pragma unroll
  for (int tm = 0; tm < 2; ++tm) {
#pragma unroll
    for (int tn = 0; tn < 2; ++tn) {
#pragma unroll
      for (int e = 0; e < 8; ++e) {
        int row = mBase + tm * 16 + e + 8 * half;
        int col = nBase + tn * 16 + l16;
        float v = acc[tm][tn][e];
        if (bias)     v += bias[col];
        if (residual) v += residual[(size_t)row * N + col];
        if (relu)     v  = fmaxf(v, 0.0f);
        if (Cf)       Cf[(size_t)row * N + col] = v;
        if (Cb) {
          if (cbT) Cb[(size_t)col * M + row] = (__bf16)v;
          else     Cb[(size_t)row * N + col] = (__bf16)v;
        }
      }
    }
  }
}

// ============================================================
// 4) flash attention, one wave per 16 query rows, per head.
//    Q  : (H, S, 64) bf16 row-major
//    K  : (H, S, 64) bf16 row-major   (used directly as B fragments)
//    Vt : (H, 64, S) bf16             (V transposed -> contiguous B fragments)
//    Zb : (S, 512)   bf16  (concat heads: head h -> cols h*64..h*64+63)
// ============================================================
__global__ void attn_kernel(const __bf16* __restrict__ Q, const __bf16* __restrict__ Kb,
                            const __bf16* __restrict__ Vt, __bf16* __restrict__ Zb) {
  int h    = blockIdx.y;
  int lane = threadIdx.x & 31;
  int wave = threadIdx.x >> 5;
  int qBase = (blockIdx.x * 8 + wave) * 16;
  int half = lane >> 4, l16 = lane & 15;
  int aOff = half ? 8 : 0;
  int bOff = half ? 8 : 0;

  const __bf16* Qh = Q  + (size_t)h * S_ * DK_;
  const __bf16* Kh = Kb + (size_t)h * S_ * DK_;
  const __bf16* Vh = Vt + (size_t)h * DK_ * S_;

  __shared__ __bf16 Plds[8][16 * 32];
  __bf16* P = Plds[wave];

  // Q fragments for this wave's 16 rows (K dim 64 -> 2 fragments), loaded once.
  v16bf qf[2];
  {
    const __bf16* qp = Qh + (size_t)(qBase + l16) * DK_;
#pragma unroll
    for (int kk = 0; kk < 2; ++kk)
      qf[kk] = cat16(ld8(qp + kk * 32 + aOff), ld8(qp + kk * 32 + aOff + 16));
  }

  float mrow[8], lrow[8], alpha[8];
  v8f o[4] = {};
#pragma unroll
  for (int v = 0; v < 8; ++v) { mrow[v] = -1e30f; lrow[v] = 0.0f; }

  const float scale = 0.125f;  // 1/sqrt(64)

  for (int jb = 0; jb < S_; jb += 32) {
    // ---- scores S(16x32) = Q(16x64) K_blk^T ----
    v8f s0 = {}, s1 = {};
#pragma unroll
    for (int kk = 0; kk < 2; ++kk) {
      const __bf16* k0p = Kh + (size_t)(jb + l16) * DK_ + kk * 32 + bOff;
      const __bf16* k1p = Kh + (size_t)(jb + 16 + l16) * DK_ + kk * 32 + bOff;
      v16bf kf0 = cat16(ld8(k0p), ld8(k0p + 16));
      v16bf kf1 = cat16(ld8(k1p), ld8(k1p + 16));
      s0 = wmma_bf16(qf[kk], kf0, s0);
      s1 = wmma_bf16(qf[kk], kf1, s1);
    }

    // ---- online softmax (per row; rows live in VGPR index within lane-half) ----
#pragma unroll
    for (int v = 0; v < 8; ++v) {
      float a = s0[v] * scale;
      float b = s1[v] * scale;
      float bm = fmaxf(a, b);
#pragma unroll
      for (int off = 1; off < 16; off <<= 1)
        bm = fmaxf(bm, __shfl_xor(bm, off, 32));
      float mn = fmaxf(mrow[v], bm);
      alpha[v] = __expf(mrow[v] - mn);
      float p0 = __expf(a - mn);
      float p1 = __expf(b - mn);
      float ps = p0 + p1;
#pragma unroll
      for (int off = 1; off < 16; off <<= 1)
        ps += __shfl_xor(ps, off, 32);
      lrow[v] = lrow[v] * alpha[v] + ps;
      mrow[v] = mn;
      int row = v + 8 * half;
      P[row * 32 + l16]      = (__bf16)p0;
      P[row * 32 + 16 + l16] = (__bf16)p1;
    }
    asm volatile("s_wait_dscnt 0" ::: "memory");  // wave-private LDS round-trip

    // ---- P fragment (A layout, 16x32) from LDS ----
    const __bf16* pp = P + l16 * 32;
    v16bf pf = cat16(ld8(pp + aOff), ld8(pp + aOff + 16));

    // ---- O(16x64) += P(16x32) V_blk(32x64), with rescale of old O ----
#pragma unroll
    for (int ft = 0; ft < 4; ++ft) {
      const __bf16* vp = Vh + (size_t)(ft * 16 + l16) * S_ + jb + bOff;
      v16bf vf = cat16(ld8(vp), ld8(vp + 16));
      v8f oc;
#pragma unroll
      for (int e = 0; e < 8; ++e) oc[e] = o[ft][e] * alpha[e];
      o[ft] = wmma_bf16(pf, vf, oc);
    }
  }

  // ---- epilogue: normalize and scatter into concatenated Z ----
#pragma unroll
  for (int ft = 0; ft < 4; ++ft) {
#pragma unroll
    for (int e = 0; e < 8; ++e) {
      int row = qBase + e + 8 * half;
      int col = h * DK_ + ft * 16 + l16;
      Zb[(size_t)row * (H_ * DK_) + col] = (__bf16)(o[ft][e] / lrow[e]);
    }
  }
}

// ============================================================
// 5) LayerNorm over rows of 512; one block (256 threads) per row.
//    Writes f32 (residual path) and optional bf16 (next GEMM operand).
// ============================================================
__global__ void layernorm_kernel(const float* __restrict__ in,
                                 const float* __restrict__ g, const float* __restrict__ b,
                                 float* __restrict__ outF, __bf16* __restrict__ outB) {
  __shared__ float sm[32];
  int row = blockIdx.x;
  int t = threadIdx.x;
  const float* x = in + (size_t)row * D_;
  float v0 = x[t], v1 = x[t + 256];

  auto blockSum = [&](float v) -> float {
    int lane = t & 31, w = t >> 5;
#pragma unroll
    for (int off = 16; off; off >>= 1) v += __shfl_xor(v, off, 32);
    if (lane == 0) sm[w] = v;
    __syncthreads();
    float r = (t < 8) ? sm[t] : 0.0f;
    if (w == 0) {
#pragma unroll
      for (int off = 4; off; off >>= 1) r += __shfl_xor(r, off, 32);
      if (lane == 0) sm[0] = r;
    }
    __syncthreads();
    r = sm[0];
    __syncthreads();
    return r;
  };

  float mean = blockSum(v0 + v1) * (1.0f / D_);
  float d0 = v0 - mean, d1 = v1 - mean;
  float var = blockSum(d0 * d0 + d1 * d1) * (1.0f / D_);
  float rstd = rsqrtf(var + EPS_);
  float y0 = d0 * rstd * g[t] + b[t];
  float y1 = d1 * rstd * g[t + 256] + b[t + 256];
  if (outF) { outF[(size_t)row * D_ + t] = y0; outF[(size_t)row * D_ + t + 256] = y1; }
  if (outB) { outB[(size_t)row * D_ + t] = (__bf16)y0; outB[(size_t)row * D_ + t + 256] = (__bf16)y1; }
}

// ============================================================
// host-side orchestration
// ============================================================
extern "C" void kernel_launch(void* const* d_in, const int* in_sizes, int n_in,
                              void* d_out, int out_size, void* d_ws, size_t ws_size,
                              hipStream_t stream) {
  const float* E   = (const float*)d_in[0];
  const float* Wq  = (const float*)d_in[1];
  const float* Wk  = (const float*)d_in[2];
  const float* Wv  = (const float*)d_in[3];
  const float* WO  = (const float*)d_in[4];
  const float* W1  = (const float*)d_in[5];
  const float* b1  = (const float*)d_in[6];
  const float* W2  = (const float*)d_in[7];
  const float* b2  = (const float*)d_in[8];
  const float* g1  = (const float*)d_in[9];
  const float* be1 = (const float*)d_in[10];
  const float* g2  = (const float*)d_in[11];
  const float* be2 = (const float*)d_in[12];
  float* out = (float*)d_out;

  // ---- workspace carve (all 256B aligned) ----
  char* w = (char*)d_ws;
  auto alloc = [&](size_t bytes) -> void* {
    void* p = (void*)w;
    w += (bytes + 255) & ~(size_t)255;
    return p;
  };
  float*  x    = (float*) alloc((size_t)S_ * D_ * 4);
  __bf16* xb   = (__bf16*)alloc((size_t)S_ * D_ * 2);
  __bf16* Wqt  = (__bf16*)alloc((size_t)H_ * DK_ * D_ * 2);
  __bf16* Wkt  = (__bf16*)alloc((size_t)H_ * DK_ * D_ * 2);
  __bf16* Wvt  = (__bf16*)alloc((size_t)H_ * DK_ * D_ * 2);
  __bf16* WOt  = (__bf16*)alloc((size_t)D_ * D_ * 2);
  __bf16* W1t  = (__bf16*)alloc((size_t)DFF_ * D_ * 2);
  __bf16* W2t  = (__bf16*)alloc((size_t)D_ * DFF_ * 2);
  __bf16* Qb   = (__bf16*)alloc((size_t)H_ * S_ * DK_ * 2);
  __bf16* Kbf  = (__bf16*)alloc((size_t)H_ * S_ * DK_ * 2);
  __bf16* Vtb  = (__bf16*)alloc((size_t)H_ * DK_ * S_ * 2);
  __bf16* Zb   = (__bf16*)alloc((size_t)S_ * D_ * 2);
  float*  pre1 = (float*) alloc((size_t)S_ * D_ * 4);
  float*  Z1   = (float*) alloc((size_t)S_ * D_ * 4);
  __bf16* Z1b  = (__bf16*)alloc((size_t)S_ * D_ * 2);
  __bf16* Hb   = (__bf16*)alloc((size_t)S_ * DFF_ * 2);
  float*  pre2 = (float*) alloc((size_t)S_ * D_ * 4);
  (void)ws_size; (void)n_in; (void)in_sizes; (void)out_size;

  // ---- 1) x = E + posenc ----
  posenc_kernel<<<(S_ * D_) / 256, 256, 0, stream>>>(E, x, xb);

  // ---- 2) weight transpose+convert to bf16 Bt layout ----
  transconv_kernel<<<dim3((D_ * DK_ + 255) / 256, 1, H_), 256, 0, stream>>>(Wq, Wqt, D_, DK_);
  transconv_kernel<<<dim3((D_ * DK_ + 255) / 256, 1, H_), 256, 0, stream>>>(Wk, Wkt, D_, DK_);
  transconv_kernel<<<dim3((D_ * DK_ + 255) / 256, 1, H_), 256, 0, stream>>>(Wv, Wvt, D_, DK_);
  transconv_kernel<<<dim3((D_ * D_ + 255) / 256, 1, 1),   256, 0, stream>>>(WO, WOt, D_, D_);
  transconv_kernel<<<dim3((D_ * DFF_ + 255) / 256, 1, 1), 256, 0, stream>>>(W1, W1t, D_, DFF_);
  transconv_kernel<<<dim3((DFF_ * D_ + 255) / 256, 1, 1), 256, 0, stream>>>(W2, W2t, DFF_, D_);

  auto gemm = [&](const __bf16* A, const __bf16* Bt, int M, int N, int K,
                  float* Cf, __bf16* Cb, int cbT,
                  const float* bias, const float* res, int relu) {
    dim3 grid(N / 32, M / 256);
    gemm_bf16_wmma<<<grid, 256, 0, stream>>>(A, Bt, M, N, K, Cf, Cb, cbT, bias, res, relu);
  };

  // ---- 3) Q/K/V projections per head (V stored transposed) ----
  for (int h = 0; h < H_; ++h) {
    gemm(xb, Wqt + (size_t)h * DK_ * D_, S_, DK_, D_,
         nullptr, Qb + (size_t)h * S_ * DK_, 0, nullptr, nullptr, 0);
    gemm(xb, Wkt + (size_t)h * DK_ * D_, S_, DK_, D_,
         nullptr, Kbf + (size_t)h * S_ * DK_, 0, nullptr, nullptr, 0);
    gemm(xb, Wvt + (size_t)h * DK_ * D_, S_, DK_, D_,
         nullptr, Vtb + (size_t)h * DK_ * S_, 1, nullptr, nullptr, 0);
  }

  // ---- 4) attention (flash, online softmax) -> Zb = concat heads ----
  attn_kernel<<<dim3(S_ / 128, H_), 256, 0, stream>>>(Qb, Kbf, Vtb, Zb);

  // ---- 5) output projection + residual, then LN1 ----
  gemm(Zb, WOt, S_, D_, D_, pre1, nullptr, 0, nullptr, x, 0);
  layernorm_kernel<<<S_, 256, 0, stream>>>(pre1, g1, be1, Z1, Z1b);

  // ---- 6) FFN: relu(Z1 W1 + b1) W2 + b2 + Z1, then LN2 -> out ----
  gemm(Z1b, W1t, S_, DFF_, D_, nullptr, Hb, 0, b1, nullptr, 1);
  gemm(Hb, W2t, S_, D_, DFF_, pre2, nullptr, 0, b2, Z1, 0);
  layernorm_kernel<<<S_, 256, 0, stream>>>(pre2, g2, be2, out, nullptr);
}
